// DCNFPN_69715909149220
// MI455X (gfx1250) — matrix operand order repaired and checked by reading, not compiled, not run on hardware
//
#include <hip/hip_runtime.h>

// ---------------- types ----------------
typedef __attribute__((ext_vector_type(16))) __bf16 v16bf;
typedef __attribute__((ext_vector_type(8)))  __bf16 v8bf;
typedef __attribute__((ext_vector_type(8)))  float  v8f;
typedef __attribute__((__vector_size__(16))) int    v4i;

__device__ __forceinline__ unsigned short f2bf(float f) {
    union { float f; unsigned u; } v; v.f = f;
    unsigned r = v.u + 0x7FFFu + ((v.u >> 16) & 1u);   // round-to-nearest-even
    return (unsigned short)(r >> 16);
}
__device__ __forceinline__ float bf2f(unsigned short u) {
    union { unsigned u; float f; } v; v.u = ((unsigned)u) << 16;
    return v.f;
}

// ---------------- CDNA5 async global->LDS (guarded; fallback = reg staging) ----
#if __has_builtin(__builtin_amdgcn_global_load_async_to_lds_b128)
#define HAVE_ASYNC_LDS 1
#endif

__device__ __forceinline__ void async_b128(const void* g, void* l) {
#if HAVE_ASYNC_LDS
    __builtin_amdgcn_global_load_async_to_lds_b128(
        (v4i __attribute__((address_space(1)))*)(void*)g,
        (v4i __attribute__((address_space(3)))*)l, 0, 0);
#else
    (void)g; (void)l;
#endif
}
__device__ __forceinline__ void wait_async0() {
#if HAVE_ASYNC_LDS
#if __has_builtin(__builtin_amdgcn_s_wait_asynccnt)
    __builtin_amdgcn_s_wait_asynccnt(0);
#else
    asm volatile("s_wait_asynccnt 0" ::: "memory");
#endif
#endif
}

// ---------------- problem constants ----------------
#define BN   4
#define CCH  256
#define HH   40
#define WW   40
#define HWP  (HH*WW)            // 1600
#define MTOT (BN*HWP)           // 6400
#define KTAP 16                 // 4x4 deform kernel
#define KDEF (CCH*KTAP)         // 4096
#define K3x3 (CCH*9)            // 2304

// ---------------- NCHW(f32) -> NHWC(bf16) tiled transpose ----------------
__global__ __launch_bounds__(256)
void nchw_to_nhwc_bf16(const float* __restrict__ in, unsigned short* __restrict__ out,
                       int C, int HW) {
    __shared__ float tile[32][33];
    int b  = blockIdx.z;
    int p0 = blockIdx.x * 32;
    int c0 = blockIdx.y * 32;
#pragma unroll
    for (int i = 0; i < 4; i++) {
        int c = c0 + threadIdx.y + i * 8;
        tile[threadIdx.y + i * 8][threadIdx.x] =
            in[((size_t)b * C + c) * HW + p0 + threadIdx.x];
    }
    __syncthreads();
#pragma unroll
    for (int i = 0; i < 4; i++) {
        int p = p0 + threadIdx.y + i * 8;
        out[((size_t)b * HW + p) * C + c0 + threadIdx.x] =
            f2bf(tile[threadIdx.x][threadIdx.y + i * 8]);
    }
}

// ---------------- weight [O][C][T] f32 -> [O][T][C] bf16 ----------------
__global__ __launch_bounds__(256)
void wt_transpose_bf16(const float* __restrict__ w, unsigned short* __restrict__ wT,
                       int O, int C, int T) {
    int idx = blockIdx.x * 256 + threadIdx.x;
    if (idx >= O * C * T) return;
    int c = idx % C;
    int t = (idx / C) % T;
    int o = idx / (C * T);
    wT[((size_t)o * T + t) * C + c] = f2bf(w[((size_t)o * C + c) * T + t]);
}

// ---------------- im2col 3x3 pad1 stride1 on NHWC bf16 ----------------
__global__ __launch_bounds__(256)
void im2col3x3(const unsigned short* __restrict__ f, unsigned short* __restrict__ col) {
    int n  = blockIdx.x;
    int c  = threadIdx.x;
    int b  = n / HWP;
    int hw = n % HWP;
    int h  = hw / WW, w = hw % WW;
    size_t ob = (size_t)n * K3x3;
#pragma unroll
    for (int t = 0; t < 9; t++) {
        int y = h + t / 3 - 1;
        int x = w + t % 3 - 1;
        unsigned short v = 0;
        if ((unsigned)y < (unsigned)HH && (unsigned)x < (unsigned)WW)
            v = f[((size_t)((b * HH + y) * WW + x)) * CCH + c];
        col[ob + (size_t)t * CCH + c] = v;
    }
}

// ---------------- modulated deformable bilinear sampling ----------------
__global__ __launch_bounds__(256)
void deform_sample(const unsigned short* __restrict__ src,
                   const float* __restrict__ om,
                   unsigned short* __restrict__ sampled,
                   int Hin, int Win, int stride, int pad, int dil) {
    __shared__ int   sIdx[WW][4];
    __shared__ float sWt [WW][4];
    int h  = blockIdx.x;
    int kk = blockIdx.y;
    int b  = blockIdx.z;
    int tid = threadIdx.x;
    if (tid < WW) {
        int w = tid;
        int n = b * HWP + h * WW + w;
        const float* omp = om + (size_t)n * 48;
        float dy = omp[2 * kk];
        float dx = omp[2 * kk + 1];
        float mk = 1.0f / (1.0f + expf(-omp[32 + kk]));
        int ky = kk >> 2, kx = kk & 3;
        float py = (float)(ky * dil + h * stride - pad) + dy;
        float px = (float)(kx * dil + w * stride - pad) + dx;
        float y0f = floorf(py), x0f = floorf(px);
        float ly = py - y0f, lx = px - x0f;
        int y0 = (int)y0f, x0 = (int)x0f;
        float wts[4] = { (1.f - ly) * (1.f - lx), (1.f - ly) * lx,
                         ly * (1.f - lx),         ly * lx };
        int ys[4] = { y0, y0, y0 + 1, y0 + 1 };
        int xs[4] = { x0, x0 + 1, x0, x0 + 1 };
#pragma unroll
        for (int q = 0; q < 4; q++) {
            bool valid = (ys[q] >= 0) && (ys[q] < Hin) && (xs[q] >= 0) && (xs[q] < Win);
            int yc = ys[q] < 0 ? 0 : (ys[q] >= Hin ? Hin - 1 : ys[q]);
            int xc = xs[q] < 0 ? 0 : (xs[q] >= Win ? Win - 1 : xs[q]);
            sIdx[w][q] = ((b * Hin + yc) * Win + xc) * CCH;
            sWt [w][q] = valid ? wts[q] * mk : 0.0f;
        }
    }
    __syncthreads();
    for (int idx = tid; idx < WW * CCH; idx += 256) {
        int w = idx >> 8;
        int c = idx & 255;
        float v = sWt[w][0] * bf2f(src[sIdx[w][0] + c])
                + sWt[w][1] * bf2f(src[sIdx[w][1] + c])
                + sWt[w][2] * bf2f(src[sIdx[w][2] + c])
                + sWt[w][3] * bf2f(src[sIdx[w][3] + c]);
        int n = b * HWP + h * WW + w;
        sampled[(size_t)n * KDEF + kk * CCH + c] = f2bf(v);
    }
}

// ---------------- direct-register WMMA GEMM (small N, com conv) ----------------
template <int WM, int WN>
__global__ __launch_bounds__(128)
void gemm_direct(const unsigned short* __restrict__ A,
                 const unsigned short* __restrict__ Bw,
                 const float* __restrict__ bias,
                 float* __restrict__ outF, int N, int K) {
    int lane = threadIdx.x & 31;
    int wave = threadIdx.x >> 5;
    int row  = lane & 15;
    bool hi  = lane >= 16;

    int m0 = blockIdx.x * (4 * 16 * WM) + wave * (16 * WM);
    int n0 = blockIdx.y * (16 * WN);

    v8f acc[WM][WN] = {};

    for (int k0 = 0; k0 < K; k0 += 32) {
        v16bf a[WM];
        v16bf bm[WN];
#pragma unroll
        for (int tm = 0; tm < WM; tm++) {
            const unsigned short* p =
                A + (size_t)(m0 + tm * 16 + row) * K + k0 + (hi ? 8 : 0);
            v8bf lo  = *(const v8bf*)(p);
            v8bf hic = *(const v8bf*)(p + 16);
            a[tm] = __builtin_shufflevector(lo, hic,
                        0, 1, 2, 3, 4, 5, 6, 7, 8, 9, 10, 11, 12, 13, 14, 15);
            __builtin_prefetch(p + 32, 0, 1);   // speculative, safe past end
        }
#pragma unroll
        for (int tn = 0; tn < WN; tn++) {
            const unsigned short* p =
                Bw + (size_t)(n0 + tn * 16 + row) * K + k0 + (hi ? 8 : 0);
            bm[tn] = __builtin_shufflevector(*(const v8bf*)(p), *(const v8bf*)(p + 16),
                        0, 1, 2, 3, 4, 5, 6, 7, 8, 9, 10, 11, 12, 13, 14, 15);
        }
#pragma unroll
        for (int tm = 0; tm < WM; tm++)
#pragma unroll
            for (int tn = 0; tn < WN; tn++)
                acc[tm][tn] = __builtin_amdgcn_wmma_f32_16x16x32_bf16(
                    false, a[tm], false, bm[tn], (short)0, acc[tm][tn], false, false);
    }

#pragma unroll
    for (int tm = 0; tm < WM; tm++) {
#pragma unroll
        for (int tn = 0; tn < WN; tn++) {
            int o = n0 + tn * 16 + row;
            float bo = bias[o];
            int mbase = m0 + tm * 16 + (hi ? 8 : 0);
#pragma unroll
            for (int j = 0; j < 8; j++)
                outF[(size_t)(mbase + j) * N + o] = acc[tm][tn][j] + bo;
        }
    }
}

// ---------------- LDS-staged WMMA GEMM (N=256 paths) ----------------
// 256 threads = 8 waves (4 along M x 2 along N); block tile 128x64, K-step 64,
// double-buffered LDS (54KB). Wave tile 32x32 -> 8 wmma per K-step.
// Staging uses CDNA5 async global->LDS (ASYNCcnt) when available.
// MODE 1: outH[m*256+o] = bf16( relu(acc+bias) + bf2f(f_old[m*256+o]) )
// MODE 2: outF[(b*256+o)*1600+hw] = acc + bias[o] + fh[same]
template <int MODE>
__global__ __launch_bounds__(256)
void gemm_lds(const unsigned short* __restrict__ A,
              const unsigned short* __restrict__ Bw,
              const float* __restrict__ bias,
              const unsigned short* __restrict__ f_old,
              const float* __restrict__ fh,
              float* __restrict__ outF,
              unsigned short* __restrict__ outH,
              int K) {
    __shared__ unsigned short Al[2][128][72];   // pad 8 halves: bank-conflict-free
    __shared__ unsigned short Bl[2][64][72];

    int tid  = threadIdx.x;
    int lane = tid & 31;
    int wave = tid >> 5;
    int waveM = wave >> 1;          // 0..3
    int waveN = wave & 1;           // 0..1
    int row  = lane & 15;
    bool hi  = lane >= 16;

    int m0blk = blockIdx.x * 128;
    int n0blk = blockIdx.y * 64;

    // staging assignment (per thread): A: 64B (row=tid/2, 2x32B), B: 32B (row=tid/4)
    int aRow = tid >> 1;
    int aCol = (tid & 1) * 32;
    int bRow = tid >> 2;
    int bCol = (tid & 3) * 16;
    const unsigned short* gA = A  + (size_t)(m0blk + aRow) * K + aCol;
    const unsigned short* gB = Bw + (size_t)(n0blk + bRow) * K + bCol;

    v8f acc[2][2] = {};
    int nk = K >> 6;                // K / 64

#if HAVE_ASYNC_LDS
    auto stage = [&](int buf, int kb) {
        const unsigned short* pA = gA + (size_t)kb * 64;
        async_b128(pA,      &Al[buf][aRow][aCol]);
        async_b128(pA + 8,  &Al[buf][aRow][aCol + 8]);
        async_b128(pA + 16, &Al[buf][aRow][aCol + 16]);
        async_b128(pA + 24, &Al[buf][aRow][aCol + 24]);
        const unsigned short* pB = gB + (size_t)kb * 64;
        async_b128(pB,     &Bl[buf][bRow][bCol]);
        async_b128(pB + 8, &Bl[buf][bRow][bCol + 8]);
    };
    stage(0, 0);
    wait_async0();
    __syncthreads();
#else
    v16bf ra0 = *(const v16bf*)(gA);
    v16bf ra1 = *(const v16bf*)(gA + 16);
    v16bf rb  = *(const v16bf*)(gB);
    *(v16bf*)&Al[0][aRow][aCol]      = ra0;
    *(v16bf*)&Al[0][aRow][aCol + 16] = ra1;
    *(v16bf*)&Bl[0][bRow][bCol]      = rb;
    __syncthreads();
#endif

    for (int kb = 0; kb < nk; kb++) {
        int cur = kb & 1;
        int nxt = cur ^ 1;
        bool more = (kb + 1) < nk;
#if HAVE_ASYNC_LDS
        if (more) stage(nxt, kb + 1);       // overlaps with compute below
#else
        if (more) {
            const unsigned short* pA = gA + (size_t)(kb + 1) * 64;
            const unsigned short* pB = gB + (size_t)(kb + 1) * 64;
            ra0 = *(const v16bf*)(pA);
            ra1 = *(const v16bf*)(pA + 16);
            rb  = *(const v16bf*)(pB);
        }
#endif
#pragma unroll
        for (int ks = 0; ks < 64; ks += 32) {
            v16bf a[2], bm[2];
#pragma unroll
            for (int tm = 0; tm < 2; tm++) {
                const unsigned short* p =
                    &Al[cur][waveM * 32 + tm * 16 + row][ks + (hi ? 8 : 0)];
                v8bf lo  = *(const v8bf*)(p);
                v8bf hic = *(const v8bf*)(p + 16);
                a[tm] = __builtin_shufflevector(lo, hic,
                            0, 1, 2, 3, 4, 5, 6, 7, 8, 9, 10, 11, 12, 13, 14, 15);
            }
#pragma unroll
            for (int tn = 0; tn < 2; tn++) {
                const unsigned short* q =
                    &Bl[cur][waveN * 32 + tn * 16 + row][ks + (hi ? 8 : 0)];
                bm[tn] = __builtin_shufflevector(*(const v8bf*)(q), *(const v8bf*)(q + 16),
                            0, 1, 2, 3, 4, 5, 6, 7, 8, 9, 10, 11, 12, 13, 14, 15);
            }
#pragma unroll
            for (int tm = 0; tm < 2; tm++)
#pragma unroll
                for (int tn = 0; tn < 2; tn++)
                    acc[tm][tn] = __builtin_amdgcn_wmma_f32_16x16x32_bf16(
                        false, a[tm], false, bm[tn], (short)0, acc[tm][tn], false, false);
        }
#if HAVE_ASYNC_LDS
        if (more) wait_async0();
#else
        if (more) {
            *(v16bf*)&Al[nxt][aRow][aCol]      = ra0;
            *(v16bf*)&Al[nxt][aRow][aCol + 16] = ra1;
            *(v16bf*)&Bl[nxt][bRow][bCol]      = rb;
        }
#endif
        __syncthreads();
    }

    int m0 = m0blk + waveM * 32;
    int n0 = n0blk + waveN * 32;
#pragma unroll
    for (int tm = 0; tm < 2; tm++) {
#pragma unroll
        for (int tn = 0; tn < 2; tn++) {
            int o = n0 + tn * 16 + row;
            float bo = bias[o];
            int mbase = m0 + tm * 16 + (hi ? 8 : 0);
#pragma unroll
            for (int j = 0; j < 8; j++) {
                int m = mbase + j;
                float v = acc[tm][tn][j] + bo;
                if (MODE == 1) {
                    size_t idx = (size_t)m * CCH + o;
                    v = fmaxf(v, 0.0f) + bf2f(f_old[idx]);
                    outH[idx] = f2bf(v);
                } else {
                    int b = m / HWP, hw = m % HWP;
                    size_t idx = ((size_t)(b * CCH + o)) * HWP + hw;
                    outF[idx] = v + fh[idx];
                }
            }
        }
    }
}

// ---------------- host orchestration ----------------
extern "C" void kernel_launch(void* const* d_in, const int* in_sizes, int n_in,
                              void* d_out, int out_size, void* d_ws, size_t ws_size,
                              hipStream_t stream) {
    (void)in_sizes; (void)n_in; (void)out_size; (void)ws_size;
    const float* f0     = (const float*)d_in[0];
    const float* f1     = (const float*)d_in[1];
    const float* f2     = (const float*)d_in[2];
    const float* com_w[2] = { (const float*)d_in[3], (const float*)d_in[5] };
    const float* com_b[2] = { (const float*)d_in[4], (const float*)d_in[6] };
    const float* dcn_w[2] = { (const float*)d_in[7], (const float*)d_in[9] };
    const float* dcn_b[2] = { (const float*)d_in[8], (const float*)d_in[10] };
    const float* res_w  = (const float*)d_in[11];
    const float* res_b  = (const float*)d_in[12];
    float* out = (float*)d_out;

    char* ws = (char*)d_ws;
    auto take = [&](size_t bytes) -> void* {
        void* p = (void*)ws;
        ws += (bytes + 255) & ~(size_t)255;
        return p;
    };
    unsigned short* f0n   = (unsigned short*)take((size_t)BN * 160 * 160 * CCH * 2);
    unsigned short* f1n   = (unsigned short*)take((size_t)BN * 80 * 80 * CCH * 2);
    unsigned short* fA    = (unsigned short*)take((size_t)MTOT * CCH * 2);
    unsigned short* fB    = (unsigned short*)take((size_t)MTOT * CCH * 2);
    unsigned short* col   = (unsigned short*)take((size_t)MTOT * K3x3 * 2);
    unsigned short* smp   = (unsigned short*)take((size_t)MTOT * KDEF * 2);
    float*          om    = (float*)take((size_t)MTOT * 48 * 4);
    unsigned short* comT[2], *dcnT[2];
    comT[0] = (unsigned short*)take((size_t)48 * K3x3 * 2);
    comT[1] = (unsigned short*)take((size_t)48 * K3x3 * 2);
    dcnT[0] = (unsigned short*)take((size_t)CCH * KDEF * 2);
    dcnT[1] = (unsigned short*)take((size_t)CCH * KDEF * 2);
    unsigned short* resT = (unsigned short*)take((size_t)CCH * K3x3 * 2);

    dim3 tb(32, 8);
    nchw_to_nhwc_bf16<<<dim3(160 * 160 / 32, CCH / 32, BN), tb, 0, stream>>>(f0, f0n, CCH, 160 * 160);
    nchw_to_nhwc_bf16<<<dim3(80 * 80 / 32, CCH / 32, BN),   tb, 0, stream>>>(f1, f1n, CCH, 80 * 80);
    nchw_to_nhwc_bf16<<<dim3(HWP / 32, CCH / 32, BN),       tb, 0, stream>>>(f2, fA, CCH, HWP);
    wt_transpose_bf16<<<(48 * CCH * 9 + 255) / 256, 256, 0, stream>>>(com_w[0], comT[0], 48, CCH, 9);
    wt_transpose_bf16<<<(48 * CCH * 9 + 255) / 256, 256, 0, stream>>>(com_w[1], comT[1], 48, CCH, 9);
    wt_transpose_bf16<<<(CCH * CCH * 16 + 255) / 256, 256, 0, stream>>>(dcn_w[0], dcnT[0], CCH, CCH, 16);
    wt_transpose_bf16<<<(CCH * CCH * 16 + 255) / 256, 256, 0, stream>>>(dcn_w[1], dcnT[1], CCH, CCH, 16);
    wt_transpose_bf16<<<(CCH * CCH * 9 + 255) / 256, 256, 0, stream>>>(res_w, resT, CCH, CCH, 9);

    const int strideL[2] = { 2, 4 };
    const int padL[2]    = { 1, 3 };
    const int dilL[2]    = { 1, 3 };
    const unsigned short* srcL[2] = { f1n, f0n };
    const int HinL[2] = { 80, 160 };

    unsigned short* fcur = fA;
    unsigned short* fnxt = fB;
    for (int it = 0; it < 4; it++) {
        int l = it & 1;
        im2col3x3<<<MTOT, 256, 0, stream>>>(fcur, col);
        gemm_direct<2, 3><<<dim3(MTOT / 128, 1), 128, 0, stream>>>(
            col, comT[l], com_b[l], om, 48, K3x3);
        deform_sample<<<dim3(HH, KTAP, BN), 256, 0, stream>>>(
            srcL[l], om, smp, HinL[l], HinL[l], strideL[l], padL[l], dilL[l]);
        gemm_lds<1><<<dim3(MTOT / 128, CCH / 64), 256, 0, stream>>>(
            smp, dcnT[l], dcn_b[l], fcur, nullptr, nullptr, fnxt, KDEF);
        unsigned short* t = fcur; fcur = fnxt; fnxt = t;
    }

    im2col3x3<<<MTOT, 256, 0, stream>>>(fcur, col);
    gemm_lds<2><<<dim3(MTOT / 128, CCH / 64), 256, 0, stream>>>(
        col, resT, res_b, nullptr, f2, out, nullptr, K3x3);
}
